// LinearAttention_65326452572467
// MI455X (gfx1250) — matrix operand order repaired and checked by reading, not compiled, not run on hardware
//
#include <hip/hip_runtime.h>
#include <math.h>

typedef __attribute__((ext_vector_type(16))) __bf16 v16bf;
typedef __attribute__((ext_vector_type(8)))  float  v8f;

#define S_LEN   2048
#define DHEAD   64
#define BH_CNT  32         // B*H = 2*16
#define NWAVE   8          // waves per block (wave32)
#define BLK_M   (16*NWAVE) // 128 queries per workgroup
#define QTILES  (S_LEN / BLK_M) // 16
#define NEG_LN10000_OVER_D (-0.14391156686f) // -ln(10000)/64

union bfpair { unsigned u; __bf16 h[2]; };

// ---- DPP16 16-lane butterfly reductions (no LDS, no waits) ----
template <int CTRL>
__device__ __forceinline__ float dpp_movf(float x) {
    return __int_as_float(__builtin_amdgcn_update_dpp(
        0, __float_as_int(x), CTRL, 0xF, 0xF, true));
}
__device__ __forceinline__ float red16_max(float x) {
    x = fmaxf(x, dpp_movf<0xB1>(x));   // quad_perm [1,0,3,2]  : xor 1
    x = fmaxf(x, dpp_movf<0x4E>(x));   // quad_perm [2,3,0,1]  : xor 2
    x = fmaxf(x, dpp_movf<0x141>(x));  // row_half_mirror      : combine quads
    x = fmaxf(x, dpp_movf<0x140>(x));  // row_mirror           : combine halves
    return x;
}
__device__ __forceinline__ float red16_sum(float x) {
    x += dpp_movf<0xB1>(x);
    x += dpp_movf<0x4E>(x);
    x += dpp_movf<0x141>(x);
    x += dpp_movf<0x140>(x);
    return x;
}

// RoPE for an (even,odd) pair at feature index d (even), position pos.
__device__ __forceinline__ void rope_pair(float xe, float xo, int pos, int d,
                                          float& re, float& ro) {
    float freq = __expf((float)d * NEG_LN10000_OVER_D);
    float ang  = (float)pos * freq;
    float sn, cs;
    __sincosf(ang, &sn, &cs);
    re = xe * cs - xo * sn;
    ro = xe * sn + xo * cs;
}

// A-matrix (16x32 bf16) half-element e (0..15) -> K index, for lane group g.
__device__ __forceinline__ int a_klocal(int e, int g) {
    return (e & 7) + ((e >> 3) << 4) + (g << 3);
}

// =====================================================================
// Pass 1: RoPE(Q), RoPE(K) -> bf16 row-major; V -> bf16 TRANSPOSED
// Vt[bh][d][s] via LDS-tiled transpose. One block per (bh, 64-row tile).
// =====================================================================
__global__ __launch_bounds__(256)
void rope_pre_tiled(const float* __restrict__ Q,
                    const float* __restrict__ K,
                    const float* __restrict__ V,
                    __bf16* __restrict__ Qr,
                    __bf16* __restrict__ Kr,
                    __bf16* __restrict__ Vt) {
    __shared__ __bf16 lv[64][72];      // 64 s-rows x 64 d (+pad)

    const int t  = threadIdx.x;
    const int bh = blockIdx.x >> 5;    // S/64 = 32 tiles per bh
    const int st = blockIdx.x & 31;
    const int s0 = st * 64;

    const size_t base = (size_t)bh * S_LEN * DHEAD;
    const int pr = t & 31;             // d-pair
    const int r0 = t >> 5;             // 0..7
    const int d  = 2 * pr;
    const float freq = __expf((float)d * NEG_LN10000_OVER_D);

#pragma unroll
    for (int i = 0; i < 8; ++i) {
        const int row = r0 + 8 * i;    // s within tile
        const int s   = s0 + row;
        const size_t off = base + (size_t)s * DHEAD + d;
        float sn, cs;
        __sincosf((float)s * freq, &sn, &cs);

        bfpair w;
        const float2 q2 = *(const float2*)(Q + off);
        w.h[0] = (__bf16)(q2.x * cs - q2.y * sn);
        w.h[1] = (__bf16)(q2.x * sn + q2.y * cs);
        *(unsigned*)(Qr + off) = w.u;

        const float2 k2 = *(const float2*)(K + off);
        w.h[0] = (__bf16)(k2.x * cs - k2.y * sn);
        w.h[1] = (__bf16)(k2.x * sn + k2.y * cs);
        *(unsigned*)(Kr + off) = w.u;

        const float2 v2 = *(const float2*)(V + off);
        lv[row][d]     = (__bf16)v2.x;
        lv[row][d + 1] = (__bf16)v2.y;
    }
    __syncthreads();

    const int sl = t & 63;             // s within tile (coalesced writes)
    const int d0 = t >> 6;             // 0..3
    const size_t vtb = (size_t)bh * DHEAD * S_LEN;
#pragma unroll
    for (int i = 0; i < 16; ++i) {
        const int dd = d0 + 4 * i;
        Vt[vtb + (size_t)dd * S_LEN + s0 + sl] = lv[sl][dd];
    }
}

// =====================================================================
// Pass 2: flash attention on pre-rotated bf16 tensors, transposed V.
// =====================================================================
__global__ __launch_bounds__(NWAVE * 32)
void attn_wmma_pre(const __bf16* __restrict__ Qr,
                   const __bf16* __restrict__ Kr,
                   const __bf16* __restrict__ Vt,
                   float* __restrict__ O) {
    __shared__ __bf16 pt[NWAVE][16][32]; // P tile staging (score->A layout)

    const int tid  = threadIdx.x;
    const int wave = tid >> 5;
    const int lane = tid & 31;
    const int lm   = lane & 15;
    const int g    = lane >> 4;

    const int bh = blockIdx.x / QTILES;
    const int qt = blockIdx.x % QTILES;
    const int q0 = qt * BLK_M + wave * 16;

    const size_t base = (size_t)bh * S_LEN * DHEAD;
    const __bf16* Qb  = Qr + base;
    const __bf16* Kb  = Kr + base;
    const __bf16* Vtb = Vt + (size_t)bh * DHEAD * S_LEN; // [d][s]
    float*        Ob  = O  + base;

    // ---- Q A-fragments: packed bf16 pair loads ----
    const int qrow = q0 + lm;
    const __bf16* Qp = Qb + (size_t)qrow * DHEAD;
    v16bf aq[2];
#pragma unroll
    for (int kc = 0; kc < 2; ++kc) {
#pragma unroll
        for (int p = 0; p < 8; ++p) {
            const int e = 2 * p;
            const int d = kc * 32 + a_klocal(e, g);
            bfpair w; w.u = *(const unsigned*)(Qp + d);
            aq[kc][e]     = w.h[0];
            aq[kc][e + 1] = w.h[1];
        }
    }

    // ---- softmax / accumulator state (C layout: lane holds 8 rows) ----
    const int row_off = g * 8;
    float mrow[8], lrow[8];
    v8f acc[4];
#pragma unroll
    for (int r = 0; r < 8; ++r) { mrow[r] = -INFINITY; lrow[r] = 0.0f; }
#pragma unroll
    for (int t = 0; t < 4; ++t)
#pragma unroll
        for (int r = 0; r < 8; ++r) acc[t][r] = 0.0f;

    const int kend = q0 + 15;

    for (int kb = 0; kb <= kend; kb += 32) {
        if (kb + 32 <= kend) { // wave-uniform; prefetch next K rows + Vt spans
            __builtin_prefetch(Kb  + (size_t)(kb + 32 + lane) * DHEAD, 0, 1);
            __builtin_prefetch(Vtb + (size_t)lane        * S_LEN + kb + 32, 0, 1);
            __builtin_prefetch(Vtb + (size_t)(lane + 32) * S_LEN + kb + 32, 0, 1);
        }

        // ---- K^T B fragments: packed pair loads (merge into b128) ----
        v16bf bk[2][2];
#pragma unroll
        for (int nt = 0; nt < 2; ++nt) {
            const int key  = kb + nt * 16 + lm;
            const int keyc = key < S_LEN ? key : S_LEN - 1;
            const __bf16* Kp = Kb + (size_t)keyc * DHEAD;
#pragma unroll
            for (int kc = 0; kc < 2; ++kc) {
#pragma unroll
                for (int p = 0; p < 8; ++p) {
                    const int e = 2 * p;
                    const int d = kc * 32 + a_klocal(e, g);
                    bfpair w; w.u = *(const unsigned*)(Kp + d);
                    bk[nt][kc][e]     = w.h[0];
                    bk[nt][kc][e + 1] = w.h[1];
                }
            }
        }

        // ---- scores: S = Qr * Kr^T ----
        v8f z = {0.f,0.f,0.f,0.f,0.f,0.f,0.f,0.f};
        v8f s0 = __builtin_amdgcn_wmma_f32_16x16x32_bf16(false, aq[0], false, bk[0][0], (short)0, z,  false, false);
        s0     = __builtin_amdgcn_wmma_f32_16x16x32_bf16(false, aq[1], false, bk[0][1], (short)0, s0, false, false);
        v8f s1 = __builtin_amdgcn_wmma_f32_16x16x32_bf16(false, aq[0], false, bk[1][0], (short)0, z,  false, false);
        s1     = __builtin_amdgcn_wmma_f32_16x16x32_bf16(false, aq[1], false, bk[1][1], (short)0, s1, false, false);

        // ---- causal mask + online softmax (DPP reductions, no LDS) ----
        const int key0 = kb + lm;
        const int key1 = kb + 16 + lm;
#pragma unroll
        for (int r = 0; r < 8; ++r) {
            const int qr = q0 + r + row_off;
            float a = (key0 <= qr) ? s0[r] : -INFINITY;
            float b = (key1 <= qr) ? s1[r] : -INFINITY;

            const float mx   = red16_max(fmaxf(a, b));
            const float mnew = fmaxf(mrow[r], mx);
            const float p0   = __expf(a - mnew);
            const float p1   = __expf(b - mnew);
            const float corr = __expf(mrow[r] - mnew); // 0 on first iter
            const float rs   = red16_sum(p0 + p1);

            lrow[r] = lrow[r] * corr + rs;
            mrow[r] = mnew;
#pragma unroll
            for (int t = 0; t < 4; ++t) acc[t][r] *= corr;

            pt[wave][r + row_off][lm]      = (__bf16)p0;
            pt[wave][r + row_off][lm + 16] = (__bf16)p1;
        }

        // wave-local DS fence (DS ops are in-order per wave)
        asm volatile("s_wait_dscnt 0x0" ::: "memory");

        // ---- reload P as A-matrix fragment ----
        v16bf pa;
#pragma unroll
        for (int p = 0; p < 8; ++p) {
            const int e  = 2 * p;
            const int kl = e + 16 * g;
            pa[e]     = pt[wave][lm][kl];
            pa[e + 1] = pt[wave][lm][kl + 1];
        }

        // ---- O += P * V : Vt gives 16 contiguous bf16 per lane ----
#pragma unroll
        for (int t = 0; t < 4; ++t) {
            const int d = 16 * t + lm;
            const v16bf bv = *(const v16bf*)(Vtb + (size_t)d * S_LEN + kb + 16 * g);
            acc[t] = __builtin_amdgcn_wmma_f32_16x16x32_bf16(false, pa, false, bv, (short)0, acc[t], false, false);
        }
    }

    // ---- epilogue ----
#pragma unroll
    for (int r = 0; r < 8; ++r) {
        const int qr = q0 + r + row_off;
        const float inv = 1.0f / lrow[r];
        float* Op = Ob + (size_t)qr * DHEAD;
#pragma unroll
        for (int t = 0; t < 4; ++t)
            Op[16 * t + lm] = acc[t][r] * inv;
    }
}

// =====================================================================
// Fallback: fully fused kernel (RoPE in-loop), used only if d_ws is too
// small for the precompute buffers. Deterministic (depends on ws_size).
// =====================================================================
__global__ __launch_bounds__(NWAVE * 32)
void attn_wmma_fused(const float* __restrict__ Q,
                     const float* __restrict__ K,
                     const float* __restrict__ V,
                     float* __restrict__ O) {
    __shared__ __bf16 pt[NWAVE][16][32];

    const int tid  = threadIdx.x;
    const int wave = tid >> 5;
    const int lane = tid & 31;
    const int lm   = lane & 15;
    const int g    = lane >> 4;

    const int bh = blockIdx.x / QTILES;
    const int qt = blockIdx.x % QTILES;
    const int q0 = qt * BLK_M + wave * 16;

    const size_t base = (size_t)bh * S_LEN * DHEAD;
    const float* Qb = Q + base;
    const float* Kb = K + base;
    const float* Vb = V + base;
    float*       Ob = O + base;

    const int qrow = q0 + lm;
    const float* Qp = Qb + (size_t)qrow * DHEAD;
    v16bf aq[2];
#pragma unroll
    for (int kc = 0; kc < 2; ++kc)
#pragma unroll
        for (int p = 0; p < 8; ++p) {
            const int e = 2 * p;
            const int d = kc * 32 + a_klocal(e, g);
            float re, ro;
            rope_pair(Qp[d], Qp[d + 1], qrow, d, re, ro);
            aq[kc][e] = (__bf16)re; aq[kc][e + 1] = (__bf16)ro;
        }

    const int row_off = g * 8;
    float mrow[8], lrow[8];
    v8f acc[4];
#pragma unroll
    for (int r = 0; r < 8; ++r) { mrow[r] = -INFINITY; lrow[r] = 0.0f; }
#pragma unroll
    for (int t = 0; t < 4; ++t)
#pragma unroll
        for (int r = 0; r < 8; ++r) acc[t][r] = 0.0f;

    const int kend = q0 + 15;
    for (int kb = 0; kb <= kend; kb += 32) {
        v16bf bk[2][2];
#pragma unroll
        for (int nt = 0; nt < 2; ++nt) {
            const int key  = kb + nt * 16 + lm;
            const int keyc = key < S_LEN ? key : S_LEN - 1;
            const float* Kp = Kb + (size_t)keyc * DHEAD;
#pragma unroll
            for (int kc = 0; kc < 2; ++kc)
#pragma unroll
                for (int p = 0; p < 8; ++p) {
                    const int e = 2 * p;
                    const int d = kc * 32 + a_klocal(e, g);
                    float re, ro;
                    rope_pair(Kp[d], Kp[d + 1], keyc, d, re, ro);
                    bk[nt][kc][e] = (__bf16)re; bk[nt][kc][e + 1] = (__bf16)ro;
                }
        }

        v8f z = {0.f,0.f,0.f,0.f,0.f,0.f,0.f,0.f};
        v8f s0 = __builtin_amdgcn_wmma_f32_16x16x32_bf16(false, aq[0], false, bk[0][0], (short)0, z,  false, false);
        s0     = __builtin_amdgcn_wmma_f32_16x16x32_bf16(false, aq[1], false, bk[0][1], (short)0, s0, false, false);
        v8f s1 = __builtin_amdgcn_wmma_f32_16x16x32_bf16(false, aq[0], false, bk[1][0], (short)0, z,  false, false);
        s1     = __builtin_amdgcn_wmma_f32_16x16x32_bf16(false, aq[1], false, bk[1][1], (short)0, s1, false, false);

        const int key0 = kb + lm;
        const int key1 = kb + 16 + lm;
#pragma unroll
        for (int r = 0; r < 8; ++r) {
            const int qr = q0 + r + row_off;
            float a = (key0 <= qr) ? s0[r] : -INFINITY;
            float b = (key1 <= qr) ? s1[r] : -INFINITY;
            const float mx   = red16_max(fmaxf(a, b));
            const float mnew = fmaxf(mrow[r], mx);
            const float p0   = __expf(a - mnew);
            const float p1   = __expf(b - mnew);
            const float corr = __expf(mrow[r] - mnew);
            const float rs   = red16_sum(p0 + p1);
            lrow[r] = lrow[r] * corr + rs;
            mrow[r] = mnew;
#pragma unroll
            for (int t = 0; t < 4; ++t) acc[t][r] *= corr;
            pt[wave][r + row_off][lm]      = (__bf16)p0;
            pt[wave][r + row_off][lm + 16] = (__bf16)p1;
        }

        asm volatile("s_wait_dscnt 0x0" ::: "memory");

        v16bf pa;
#pragma unroll
        for (int p = 0; p < 8; ++p) {
            const int e  = 2 * p;
            const int kl = e + 16 * g;
            pa[e] = pt[wave][lm][kl]; pa[e + 1] = pt[wave][lm][kl + 1];
        }

#pragma unroll
        for (int t = 0; t < 4; ++t) {
            const int d = 16 * t + lm;
            v16bf bv;
#pragma unroll
            for (int p = 0; p < 8; ++p) {
                const int e = 2 * p;
                int k0 = kb + e + 16 * g;
                int k1 = k0 + 1;
                k0 = k0 < S_LEN ? k0 : S_LEN - 1;
                k1 = k1 < S_LEN ? k1 : S_LEN - 1;
                bv[e]     = (__bf16)Vb[(size_t)k0 * DHEAD + d];
                bv[e + 1] = (__bf16)Vb[(size_t)k1 * DHEAD + d];
            }
            acc[t] = __builtin_amdgcn_wmma_f32_16x16x32_bf16(false, pa, false, bv, (short)0, acc[t], false, false);
        }
    }

#pragma unroll
    for (int r = 0; r < 8; ++r) {
        const int qr = q0 + r + row_off;
        const float inv = 1.0f / lrow[r];
        float* Op = Ob + (size_t)qr * DHEAD;
#pragma unroll
        for (int t = 0; t < 4; ++t)
            Op[16 * t + lm] = acc[t][r] * inv;
    }
}

extern "C" void kernel_launch(void* const* d_in, const int* in_sizes, int n_in,
                              void* d_out, int out_size, void* d_ws, size_t ws_size,
                              hipStream_t stream) {
    (void)in_sizes; (void)n_in; (void)out_size;
    const float* Q = (const float*)d_in[0];
    const float* K = (const float*)d_in[1];
    const float* V = (const float*)d_in[2];
    float* O = (float*)d_out;

    const size_t elems = (size_t)BH_CNT * S_LEN * DHEAD; // 4,194,304
    const size_t need  = 3 * elems * 2 + 128;            // 3 bf16 tensors + tail pad

    dim3 agrid(BH_CNT * QTILES);   // 512 workgroups
    dim3 ablock(NWAVE * 32);       // 256 threads = 8 waves

    if (ws_size >= need && d_ws != nullptr) {
        __bf16* Qr = (__bf16*)d_ws;
        __bf16* Kr = Qr + elems;
        __bf16* Vt = Kr + elems;   // transposed [bh][d][s]
        rope_pre_tiled<<<dim3(BH_CNT * (S_LEN / 64)), dim3(256), 0, stream>>>(Q, K, V, Qr, Kr, Vt);
        attn_wmma_pre<<<agrid, ablock, 0, stream>>>(Qr, Kr, Vt, O);
    } else {
        attn_wmma_fused<<<agrid, ablock, 0, stream>>>(Q, K, V, O);
    }
}
